// MultiHeadAttention_3745211482972
// MI455X (gfx1250) — matrix operand order repaired and checked by reading, not compiled, not run on hardware
//
#include <hip/hip_runtime.h>
#include <hip/hip_bf16.h>

// ---------------------------------------------------------------------------
// MHA forward for MI455X (gfx1250): bf16 WMMA GEMMs with fp32 accumulation.
// ---------------------------------------------------------------------------

typedef __bf16 bf16;
typedef __attribute__((ext_vector_type(16))) __bf16 v16bf;
typedef __attribute__((ext_vector_type(8)))  __bf16 v8bf;
typedef __attribute__((ext_vector_type(8)))  float  v8f;

#define B_    2
#define S_    2048
#define H_    1024
#define NH_   16
#define HD_   64
#define MROWS (B_ * S_)        // 4096
#define NEG_  -1e9f
#define EPS_  1e-6f

// ---------------------------------------------------------------------------
// WMMA fragment loaders (CDNA5 wave32 layouts, CDNA5 ISA §7.12.2)
//
// A 16x32 bf16 (M x K): lane&15 = row M; lane>=16 shifts K by 8 within each
// 16-wide half. Per-lane vector = two contiguous 8-elem (16B) chunks at
// K-offsets {8*(lane>>4), 16 + 8*(lane>>4)}.
//
// B 32x16 bf16 (K x N): lane&15 = col N; K split as for A (lane>=16 shifts K by 8).
// Per-lane vector = two 8-elem (16B) chunks of column N at K-offsets {8*(lane>>4),
// 16 + 8*(lane>>4)}; the source matrix stores B-columns as contiguous rows.
// ---------------------------------------------------------------------------

__device__ __forceinline__ v16bf load_a_bf16(const bf16* __restrict__ base,
                                             int ld, int lane) {
  const int row = lane & 15;
  const int hi  = lane >> 4;
  const bf16* p = base + (size_t)row * ld + hi * 8;
  v8bf lo = *(const v8bf*)(p);        // K = k0 + 8*hi      .. +7
  v8bf hh = *(const v8bf*)(p + 16);   // K = k0 + 16 + 8*hi .. +7
  return __builtin_shufflevector(lo, hh, 0, 1, 2, 3, 4, 5, 6, 7,
                                         8, 9, 10, 11, 12, 13, 14, 15);
}

// A fragment sourced from fp32 (attention probabilities), cvt to bf16 in regs.
__device__ __forceinline__ v16bf load_a_f32(const float* __restrict__ base,
                                            int ld, int lane) {
  const int row = lane & 15;
  const int hi  = lane >> 4;
  const float* p = base + (size_t)row * ld + hi * 8;
  v16bf a;
#pragma unroll
  for (int i = 0; i < 8; ++i) a[i] = (__bf16)p[i];
#pragma unroll
  for (int i = 0; i < 8; ++i) a[8 + i] = (__bf16)p[16 + i];
  return a;
}

// base must point at row 0 of a matrix whose row n holds B-column n (K-contig)
__device__ __forceinline__ v16bf load_b_bf16(const bf16* __restrict__ base,
                                             int ld, int lane) {
  const int col = lane & 15;
  const int hi  = lane >> 4;
  const bf16* p = base + (size_t)col * ld + hi * 8;
  v8bf lo = *(const v8bf*)(p);
  v8bf hh = *(const v8bf*)(p + 16);
  return __builtin_shufflevector(lo, hh, 0, 1, 2, 3, 4, 5, 6, 7,
                                         8, 9, 10, 11, 12, 13, 14, 15);
}

__device__ __forceinline__ v8f wmma_bf16(v16bf a, v16bf b, v8f c) {
  // emits v_wmma_f32_16x16x32_bf16
  return __builtin_amdgcn_wmma_f32_16x16x32_bf16(
      /*neg_a=*/false, a, /*neg_b=*/false, b,
      /*c_mod=*/(short)0, c, /*reuse_a=*/false, /*reuse_b=*/false);
}

// ---------------------------------------------------------------------------
// Block reductions (wave32): warp shuffle + tiny LDS combine. blockDim == 256.
// ---------------------------------------------------------------------------

__device__ __forceinline__ float block_reduce_sum(float v, float* sbuf) {
#pragma unroll
  for (int o = 16; o > 0; o >>= 1) v += __shfl_xor(v, o, 32);
  const int lane = threadIdx.x & 31, w = threadIdx.x >> 5;
  __syncthreads();
  if (lane == 0) sbuf[w] = v;
  __syncthreads();
  v = (lane < 8) ? sbuf[lane] : 0.0f;
#pragma unroll
  for (int o = 4; o > 0; o >>= 1) v += __shfl_xor(v, o, 32);
  if (threadIdx.x == 0) sbuf[0] = v;
  __syncthreads();
  return sbuf[0];
}

__device__ __forceinline__ float block_reduce_max(float v, float* sbuf) {
#pragma unroll
  for (int o = 16; o > 0; o >>= 1) v = fmaxf(v, __shfl_xor(v, o, 32));
  const int lane = threadIdx.x & 31, w = threadIdx.x >> 5;
  __syncthreads();
  if (lane == 0) sbuf[w] = v;
  __syncthreads();
  v = (lane < 8) ? sbuf[lane] : -3.4e38f;
#pragma unroll
  for (int o = 4; o > 0; o >>= 1) v = fmaxf(v, __shfl_xor(v, o, 32));
  if (threadIdx.x == 0) sbuf[0] = v;
  __syncthreads();
  return sbuf[0];
}

// ---------------------------------------------------------------------------
// Kernel 0: fp32 -> bf16 conversion
// ---------------------------------------------------------------------------
__global__ __launch_bounds__(256) void cvt_f32_bf16_kernel(
    const float* __restrict__ in, bf16* __restrict__ out, int n) {
  int i = blockIdx.x * 256 + threadIdx.x;
  if (i < n) out[i] = (__bf16)in[i];
}

// ---------------------------------------------------------------------------
// Kernel 1: QKV projections.  X[4096,1024](bf16) @ W.T for W in {Wq,Wk,Wv}.
// Block = 8 waves as 4x2, wave tile = 32x32 (2x2 WMMA, reuses A/B frags).
// Block tile 128x64.  Grid (32, 16, 3).
// Q stored * (1/sqrt(HD)) in [B,NH,S,HD]; K in [B,NH,S,HD]; V transposed
// into [B,NH,HD,S] so the PV GEMM gets contiguous B-fragments.
// ---------------------------------------------------------------------------
__global__ __launch_bounds__(256) void qkv_gemm_kernel(
    const bf16* __restrict__ X, const bf16* __restrict__ Wq,
    const bf16* __restrict__ Wk, const bf16* __restrict__ Wv,
    bf16* __restrict__ Q, bf16* __restrict__ Ko, bf16* __restrict__ Vt) {
  const int lane = threadIdx.x & 31;
  const int wave = threadIdx.x >> 5;
  const int wm = wave >> 1, wn = wave & 1;
  const int mat = blockIdx.z;
  const bf16* W = (mat == 0) ? Wq : (mat == 1) ? Wk : Wv;

  const int m0 = blockIdx.x * 128 + wm * 32;
  const int n0 = blockIdx.y * 64 + wn * 32;

  v8f acc00 = {}, acc01 = {}, acc10 = {}, acc11 = {};
  for (int k0 = 0; k0 < H_; k0 += 32) {
    if (k0 + 32 < H_) {
      __builtin_prefetch(X + (size_t)m0 * H_ + k0 + 32, 0, 1);
      __builtin_prefetch(W + (size_t)n0 * H_ + k0 + 32, 0, 1);
    }
    v16bf a0 = load_a_bf16(X + (size_t)m0 * H_ + k0, H_, lane);
    v16bf a1 = load_a_bf16(X + (size_t)(m0 + 16) * H_ + k0, H_, lane);
    v16bf b0 = load_b_bf16(W + (size_t)n0 * H_ + k0, H_, lane);
    v16bf b1 = load_b_bf16(W + (size_t)(n0 + 16) * H_ + k0, H_, lane);
    acc00 = wmma_bf16(a0, b0, acc00);
    acc01 = wmma_bf16(a0, b1, acc01);
    acc10 = wmma_bf16(a1, b0, acc10);
    acc11 = wmma_bf16(a1, b1, acc11);
  }

  const int rbase = (lane >> 4) * 8;
  const int lane15 = lane & 15;
#pragma unroll
  for (int t = 0; t < 4; ++t) {
    const v8f acc = (t == 0) ? acc00 : (t == 1) ? acc01 : (t == 2) ? acc10 : acc11;
    const int mt = m0 + ((t >> 1) ? 16 : 0);
    const int nt = n0 + ((t & 1) ? 16 : 0);
#pragma unroll
    for (int r = 0; r < 8; ++r) {
      const int m = mt + rbase + r;      // 0..4095  -> (b, s)
      const int n = nt + lane15;         // 0..1023  -> (h, d)
      const int bb = m >> 11, s = m & (S_ - 1);
      const int h = n >> 6, d = n & (HD_ - 1);
      const float v = acc[r];
      if (mat == 0)
        Q[(((size_t)bb * NH_ + h) * S_ + s) * HD_ + d] = (__bf16)(v * 0.125f);
      else if (mat == 1)
        Ko[(((size_t)bb * NH_ + h) * S_ + s) * HD_ + d] = (__bf16)v;
      else
        Vt[(((size_t)bb * NH_ + h) * HD_ + d) * S_ + s] = (__bf16)v;
    }
  }
}

// ---------------------------------------------------------------------------
// Kernel 2: scores = (Q/sqrt(HD)) @ K^T, masked, fp32 into d_out attn region.
// Per (b,h): [2048,2048] = [2048,64] x [64,2048].  Wave = one 16x16 tile,
// block 8 waves as 4x2 -> 64x32 tile.  Grid (32, 64, 32).
// ---------------------------------------------------------------------------
__global__ __launch_bounds__(256) void scores_kernel(
    const bf16* __restrict__ Q, const bf16* __restrict__ Ko,
    const float* __restrict__ mask, float* __restrict__ attn) {
  const int lane = threadIdx.x & 31;
  const int wave = threadIdx.x >> 5;
  const int wm = wave >> 1, wn = wave & 1;
  const int bh = blockIdx.z;
  const int b = bh >> 4;

  const int q0 = blockIdx.x * 64 + wm * 16;
  const int c0 = blockIdx.y * 32 + wn * 16;
  const bf16* Qb = Q + (size_t)bh * S_ * HD_;
  const bf16* Kb = Ko + (size_t)bh * S_ * HD_;

  v8f acc = {};
#pragma unroll
  for (int k0 = 0; k0 < HD_; k0 += 32) {
    v16bf a = load_a_bf16(Qb + (size_t)q0 * HD_ + k0, HD_, lane);
    v16bf bm = load_b_bf16(Kb + (size_t)c0 * HD_ + k0, HD_, lane);
    acc = wmma_bf16(a, bm, acc);
  }

  const int rbase = (lane >> 4) * 8;
  const int lane15 = lane & 15;
  const float* mb = mask + (size_t)b * S_ * S_;   // mask[b,0,:,:]
#pragma unroll
  for (int r = 0; r < 8; ++r) {
    const int q = q0 + rbase + r;
    const int c = c0 + lane15;
    const float mval = mb[(size_t)q * S_ + c];
    attn[((size_t)bh * S_ + q) * S_ + c] = (mval == 0.0f) ? NEG_ : acc[r];
  }
}

// ---------------------------------------------------------------------------
// Kernel 3: row softmax in-place on attn (B*NH*S = 65536 rows of 2048).
// ---------------------------------------------------------------------------
__global__ __launch_bounds__(256) void softmax_kernel(float* __restrict__ attn) {
  __shared__ float sred[8];
  float* p = attn + (size_t)blockIdx.x * S_;

  float mx = -3.4e38f;
  for (int i = threadIdx.x; i < S_; i += 256) mx = fmaxf(mx, p[i]);
  mx = block_reduce_max(mx, sred);

  float sum = 0.0f;
  for (int i = threadIdx.x; i < S_; i += 256) {
    const float e = __expf(p[i] - mx);
    p[i] = e;
    sum += e;
  }
  sum = block_reduce_sum(sum, sred);

  const float inv = 1.0f / sum;
  for (int i = threadIdx.x; i < S_; i += 256) p[i] *= inv;
}

// ---------------------------------------------------------------------------
// Kernel 4: O = P @ V per (b,h): [2048,64] = [2048,2048] x [2048,64].
// P read fp32 -> bf16 in regs; V read from transposed layout (contiguous).
// Block 8 waves as 2x4 -> 32(M) x 64(N=HD).  Grid (64, 1, 32).
// Output merged-heads bf16 [B,S,H].
// ---------------------------------------------------------------------------
__global__ __launch_bounds__(256) void pv_kernel(
    const float* __restrict__ attn, const bf16* __restrict__ Vt,
    bf16* __restrict__ Om) {
  const int lane = threadIdx.x & 31;
  const int wave = threadIdx.x >> 5;
  const int wm = wave >> 2, wn = wave & 3;
  const int bh = blockIdx.z;
  const int b = bh >> 4, h = bh & 15;

  const int m0 = blockIdx.x * 32 + wm * 16;   // query row s
  const int n0 = wn * 16;                     // head dim d
  const float* P = attn + (size_t)bh * S_ * S_;
  const bf16* Vb = Vt + (size_t)bh * HD_ * S_;

  v8f acc = {};
  for (int k0 = 0; k0 < S_; k0 += 32) {
    if (k0 + 32 < S_) __builtin_prefetch(P + (size_t)m0 * S_ + k0 + 32, 0, 1);
    v16bf a = load_a_f32(P + (size_t)m0 * S_ + k0, S_, lane);
    v16bf bm = load_b_bf16(Vb + (size_t)n0 * S_ + k0, S_, lane);
    acc = wmma_bf16(a, bm, acc);
  }

  const int rbase = (lane >> 4) * 8;
  const int lane15 = lane & 15;
#pragma unroll
  for (int r = 0; r < 8; ++r) {
    const int s = m0 + rbase + r;
    const int d = n0 + lane15;
    Om[((size_t)b * S_ + s) * H_ + h * HD_ + d] = (__bf16)acc[r];
  }
}

// ---------------------------------------------------------------------------
// Kernel 5: X = O @ Wo^T + enc (fp32).  Same tiling as QKV GEMM. Grid (32,16).
// ---------------------------------------------------------------------------
__global__ __launch_bounds__(256) void oproj_kernel(
    const bf16* __restrict__ O, const bf16* __restrict__ Wo,
    const float* __restrict__ enc, float* __restrict__ X) {
  const int lane = threadIdx.x & 31;
  const int wave = threadIdx.x >> 5;
  const int wm = wave >> 1, wn = wave & 1;

  const int m0 = blockIdx.x * 128 + wm * 32;
  const int n0 = blockIdx.y * 64 + wn * 32;

  v8f acc00 = {}, acc01 = {}, acc10 = {}, acc11 = {};
  for (int k0 = 0; k0 < H_; k0 += 32) {
    if (k0 + 32 < H_) {
      __builtin_prefetch(O + (size_t)m0 * H_ + k0 + 32, 0, 1);
      __builtin_prefetch(Wo + (size_t)n0 * H_ + k0 + 32, 0, 1);
    }
    v16bf a0 = load_a_bf16(O + (size_t)m0 * H_ + k0, H_, lane);
    v16bf a1 = load_a_bf16(O + (size_t)(m0 + 16) * H_ + k0, H_, lane);
    v16bf b0 = load_b_bf16(Wo + (size_t)n0 * H_ + k0, H_, lane);
    v16bf b1 = load_b_bf16(Wo + (size_t)(n0 + 16) * H_ + k0, H_, lane);
    acc00 = wmma_bf16(a0, b0, acc00);
    acc01 = wmma_bf16(a0, b1, acc01);
    acc10 = wmma_bf16(a1, b0, acc10);
    acc11 = wmma_bf16(a1, b1, acc11);
  }

  const int rbase = (lane >> 4) * 8;
  const int lane15 = lane & 15;
#pragma unroll
  for (int t = 0; t < 4; ++t) {
    const v8f acc = (t == 0) ? acc00 : (t == 1) ? acc01 : (t == 2) ? acc10 : acc11;
    const int mt = m0 + ((t >> 1) ? 16 : 0);
    const int nt = n0 + ((t & 1) ? 16 : 0);
#pragma unroll
    for (int r = 0; r < 8; ++r) {
      const size_t idx = (size_t)(mt + rbase + r) * H_ + (nt + lane15);
      X[idx] = acc[r] + enc[idx];
    }
  }
}

// ---------------------------------------------------------------------------
// Kernel 6: LayerNorm rows of 1024 -> y (fp32).  Grid 4096 x 256.
// ---------------------------------------------------------------------------
__global__ __launch_bounds__(256) void layernorm_kernel(
    const float* __restrict__ X, const float* __restrict__ w,
    const float* __restrict__ bia, float* __restrict__ Y) {
  __shared__ float sred[8];
  const float* x = X + (size_t)blockIdx.x * H_;
  float* y = Y + (size_t)blockIdx.x * H_;

  float s = 0.0f;
  for (int i = threadIdx.x; i < H_; i += 256) s += x[i];
  const float mean = block_reduce_sum(s, sred) * (1.0f / H_);

  float v = 0.0f;
  for (int i = threadIdx.x; i < H_; i += 256) {
    const float d = x[i] - mean;
    v += d * d;
  }
  const float var = block_reduce_sum(v, sred) * (1.0f / H_);
  const float inv = rsqrtf(var + EPS_);

  for (int i = threadIdx.x; i < H_; i += 256)
    y[i] = (x[i] - mean) * inv * w[i] + bia[i];
}

// ---------------------------------------------------------------------------
// Host launcher
// ---------------------------------------------------------------------------
extern "C" void kernel_launch(void* const* d_in, const int* in_sizes, int n_in,
                              void* d_out, int out_size, void* d_ws,
                              size_t ws_size, hipStream_t stream) {
  (void)in_sizes; (void)n_in; (void)out_size; (void)ws_size;

  const float* enc  = (const float*)d_in[0];
  const float* mask = (const float*)d_in[1];
  const float* Wq   = (const float*)d_in[2];
  const float* Wk   = (const float*)d_in[3];
  const float* Wv   = (const float*)d_in[4];
  const float* Wo   = (const float*)d_in[5];
  const float* lnw  = (const float*)d_in[6];
  const float* lnb  = (const float*)d_in[7];

  float* y    = (float*)d_out;                                  // [B,S,H]
  float* attn = (float*)d_out + (size_t)B_ * S_ * H_;           // [B,NH,S,S]

  // workspace carve-up (~67 MB)
  char* ws = (char*)d_ws;
  bf16* encb = (bf16*)ws; ws += (size_t)MROWS * H_ * sizeof(bf16);
  bf16* wqb  = (bf16*)ws; ws += (size_t)H_ * H_ * sizeof(bf16);
  bf16* wkb  = (bf16*)ws; ws += (size_t)H_ * H_ * sizeof(bf16);
  bf16* wvb  = (bf16*)ws; ws += (size_t)H_ * H_ * sizeof(bf16);
  bf16* wob  = (bf16*)ws; ws += (size_t)H_ * H_ * sizeof(bf16);
  bf16* Qb   = (bf16*)ws; ws += (size_t)B_ * NH_ * S_ * HD_ * sizeof(bf16);
  bf16* Kb   = (bf16*)ws; ws += (size_t)B_ * NH_ * S_ * HD_ * sizeof(bf16);
  bf16* Vtb  = (bf16*)ws; ws += (size_t)B_ * NH_ * S_ * HD_ * sizeof(bf16);
  bf16* Ob   = (bf16*)ws; ws += (size_t)MROWS * H_ * sizeof(bf16);
  float* Xb  = (float*)ws; ws += (size_t)MROWS * H_ * sizeof(float);

  int n = MROWS * H_;
  cvt_f32_bf16_kernel<<<(n + 255) / 256, 256, 0, stream>>>(enc, encb, n);
  n = H_ * H_;
  cvt_f32_bf16_kernel<<<(n + 255) / 256, 256, 0, stream>>>(Wq, wqb, n);
  cvt_f32_bf16_kernel<<<(n + 255) / 256, 256, 0, stream>>>(Wk, wkb, n);
  cvt_f32_bf16_kernel<<<(n + 255) / 256, 256, 0, stream>>>(Wv, wvb, n);
  cvt_f32_bf16_kernel<<<(n + 255) / 256, 256, 0, stream>>>(Wo, wob, n);

  qkv_gemm_kernel<<<dim3(32, 16, 3), 256, 0, stream>>>(encb, wqb, wkb, wvb,
                                                       Qb, Kb, Vtb);
  scores_kernel<<<dim3(32, 64, 32), 256, 0, stream>>>(Qb, Kb, mask, attn);
  softmax_kernel<<<dim3(B_ * NH_ * S_), 256, 0, stream>>>(attn);
  pv_kernel<<<dim3(64, 1, 32), 256, 0, stream>>>(attn, Vtb, Ob);
  oproj_kernel<<<dim3(32, 16, 1), 256, 0, stream>>>(Ob, wob, enc, Xb);
  layernorm_kernel<<<dim3(MROWS), 256, 0, stream>>>(Xb, lnw, lnb, y);
}
